// AqtConvBlock_12549894439421
// MI455X (gfx1250) — matrix operand order repaired and checked by reading, not compiled, not run on hardware
//
#include <hip/hip_runtime.h>
#include <stdint.h>

typedef __attribute__((ext_vector_type(16))) _Float16 v16h;
typedef __attribute__((ext_vector_type(8)))  _Float16 v8h;
typedef __attribute__((ext_vector_type(4)))  _Float16 v4h;
typedef __attribute__((ext_vector_type(8)))  float    v8f;

// Problem constants
#define NN   32
#define HH   112
#define WW   112
#define CIN  128
#define COUT 256
#define KTOT (9 * CIN)          // 1152
#define NHW  (NN * HH * WW)     // 401408
#define XQ_ELEMS   ((size_t)NHW * CIN)          // 51,380,224 f16
#define KQ_ELEMS   ((size_t)KTOT * COUT)        // 294,912 f16
#define XQ_BYTES   (XQ_ELEMS * 2)               // 102,760,448
#define KQ_BYTES   (KQ_ELEMS * 2)

// ---------------- stats init ----------------
__global__ void init_stats_kernel(unsigned* amaxx, unsigned* amaxk,
                                  float* gsum, float* gsumsq) {
    int t = threadIdx.x;
    if (t == 0) { *amaxx = 0u; *amaxk = 0u; }
    if (t < COUT) { gsum[t] = 0.f; gsumsq[t] = 0.f; }
}

// ---------------- abs-max reduction (per-tensor scale) ----------------
__global__ void absmax_kernel(const float* __restrict__ in, int n, unsigned* out) {
    __shared__ float red[256];
    float m = 0.f;
    for (int i = blockIdx.x * blockDim.x + threadIdx.x; i < n;
         i += gridDim.x * blockDim.x)
        m = fmaxf(m, fabsf(in[i]));
    red[threadIdx.x] = m;
    __syncthreads();
    for (int s = 128; s > 0; s >>= 1) {
        if (threadIdx.x < (unsigned)s)
            red[threadIdx.x] = fmaxf(red[threadIdx.x], red[threadIdx.x + s]);
        __syncthreads();
    }
    if (threadIdx.x == 0)
        atomicMax(out, __float_as_uint(red[0]));  // non-negative floats: uint order == float order
}

// ---------------- quantize x -> f16 (NHWC, c fastest) ----------------
__global__ void quant_x_kernel(const float4* __restrict__ x, v4h* __restrict__ xq,
                               const unsigned* __restrict__ amax, int n4) {
    float xs = 127.f / fmaxf(__uint_as_float(*amax), 1e-6f);
    for (int i = blockIdx.x * blockDim.x + threadIdx.x; i < n4;
         i += gridDim.x * blockDim.x) {
        float4 v = x[i];
        v4h h;
        h[0] = (_Float16)(v.x * xs);
        h[1] = (_Float16)(v.y * xs);
        h[2] = (_Float16)(v.z * xs);
        h[3] = (_Float16)(v.w * xs);
        xq[i] = h;
    }
}

// ---------------- quantize kernel -> f16, transposed to [COUT][K] ----------------
__global__ void quant_kT_kernel(const float* __restrict__ k, _Float16* __restrict__ kqT,
                                const unsigned* __restrict__ amax) {
    float ks = 127.f / fmaxf(__uint_as_float(*amax), 1e-6f);
    for (int i = blockIdx.x * blockDim.x + threadIdx.x; i < KTOT * COUT;
         i += gridDim.x * blockDim.x) {
        int cout = i / KTOT;
        int kk   = i - cout * KTOT;
        kqT[i] = (_Float16)(k[(size_t)kk * COUT + cout] * ks);
    }
}

// ---------------- main conv: implicit GEMM with v_wmma_f32_16x16x32_f16 -----------
// Block tile: 64 output pixels (M) x 256 cout (N). 8 waves = 2(M) x 4(N);
// each wave owns 32x64 = 2x4 WMMA accumulators. K tiled by 32 (36 chunks),
// each chunk lies inside exactly one 3x3 tap -> clean im2col addressing.
// A/B tiles staged via CDNA5 async global->LDS (ASYNCcnt), double buffered.
#define LDA 40   // padded row length (f16) -> conflict-free ds reads
#define LDB 40

__global__ __launch_bounds__(256) void conv_wmma_kernel(
        const _Float16* __restrict__ xq, const _Float16* __restrict__ kqT,
        float* __restrict__ y, float* __restrict__ gsum, float* __restrict__ gsumsq) {
    __shared__ _Float16 sA[2][64 * LDA];
    __shared__ _Float16 sB[2][COUT * LDB];
    __shared__ float ssum[COUT], ssumsq[COUT];

    const int tid   = threadIdx.x;
    const int pBase = blockIdx.x * 64;

    if (tid < COUT) { ssum[tid] = 0.f; ssumsq[tid] = 0.f; }

    // per-thread A-load coordinates (fixed across K chunks except tap offset)
    const int arow  = tid >> 2;             // 0..63
    const int apart = (tid & 3) * 8;        // f16 offset within 32-wide K chunk
    const int p     = pBase + arow;
    const int pn    = p / (HH * WW);
    const int prem  = p - pn * (HH * WW);
    const int oh    = prem / WW;
    const int ow    = prem - oh * WW;

    auto loadTile = [&](int buf, int ch) {
        const int kstart = ch * 32;
        const int tap    = kstart >> 7;      // /CIN
        const int c0     = kstart & (CIN - 1);
        const int ky     = tap / 3, kx = tap - ky * 3;
        const int ih     = oh + ky - 1, iw = ow + kx - 1;
        // A tile: 64 rows x 32 f16 (one 16B chunk per thread)
        uint32_t aoff = (uint32_t)(uintptr_t)(&sA[buf][arow * LDA + apart]);
        if ((unsigned)ih < (unsigned)HH && (unsigned)iw < (unsigned)WW) {
            const _Float16* g = xq +
                (((size_t)(pn * HH + ih) * WW + iw) * CIN + c0 + apart);
            asm volatile("global_load_async_to_lds_b128 %0, %1, off"
                         :: "v"(aoff), "v"(g) : "memory");
        } else {
            v8h z = {};
            *(v8h*)(&sA[buf][arow * LDA + apart]) = z;   // SAME-padding zero fill
        }
        // B tile (transposed weights): 256 rows x 32 f16 -> 4 chunks/thread
#pragma unroll
        for (int q = 0; q < 4; ++q) {
            int cidx  = tid + q * 256;
            int brow  = cidx >> 2;           // cout row 0..255
            int bpart = (cidx & 3) * 8;
            uint32_t boff = (uint32_t)(uintptr_t)(&sB[buf][brow * LDB + bpart]);
            const _Float16* g = kqT + (size_t)brow * KTOT + kstart + bpart;
            asm volatile("global_load_async_to_lds_b128 %0, %1, off"
                         :: "v"(boff), "v"(g) : "memory");
        }
    };

    const int wv    = tid >> 5;
    const int lane  = tid & 31;
    const int wm    = wv >> 2;       // 0..1  (M)
    const int wn    = wv & 3;        // 0..3  (N)
    const int lmod  = lane & 15;
    const int lhalf = lane >> 4;

    v8f acc[2][4] = {};

    loadTile(0, 0);
    for (int ch = 0; ch < 36; ++ch) {
        asm volatile("s_wait_asynccnt 0" ::: "memory");
        __syncthreads();
        const int cur = ch & 1;
        if (ch + 1 < 36) loadTile(cur ^ 1, ch + 1);

        // A fragments: ISA layout — lanes lo hold K{0..7,16..23}, hi K{8..15,24..31}
        v16h afrag[2];
#pragma unroll
        for (int mi = 0; mi < 2; ++mi) {
            const _Float16* ap = &sA[cur][(wm * 32 + mi * 16 + lmod) * LDA];
            v8h lo = *(const v8h*)(ap + lhalf * 8);
            v8h hi = *(const v8h*)(ap + 16 + lhalf * 8);
            afrag[mi] = __builtin_shufflevector(lo, hi, 0,1,2,3,4,5,6,7,
                                                        8,9,10,11,12,13,14,15);
        }
        // B fragments: ISA layout — lanes lo hold K{0..7,16..23}, hi K{8..15,24..31}
        v16h bfrag[4];
#pragma unroll
        for (int ni = 0; ni < 4; ++ni) {
            const _Float16* bp = &sB[cur][(wn * 64 + ni * 16 + lmod) * LDB + lhalf * 8];
            v8h lo = *(const v8h*)(bp);
            v8h hi = *(const v8h*)(bp + 16);
            bfrag[ni] = __builtin_shufflevector(lo, hi, 0,1,2,3,4,5,6,7,
                                                        8,9,10,11,12,13,14,15);
        }
#pragma unroll
        for (int mi = 0; mi < 2; ++mi)
#pragma unroll
            for (int ni = 0; ni < 4; ++ni)
                acc[mi][ni] = __builtin_amdgcn_wmma_f32_16x16x32_f16(
                    false, afrag[mi], false, bfrag[ni],
                    (short)0, acc[mi][ni], false, false);
    }

    // Epilogue: write pre-BN y (f32) and accumulate per-channel sum/sum^2
#pragma unroll
    for (int ni = 0; ni < 4; ++ni) {
        const int cl = wn * 64 + ni * 16 + lmod;        // cout 0..255
        float s = 0.f, s2 = 0.f;
#pragma unroll
        for (int mi = 0; mi < 2; ++mi) {
#pragma unroll
            for (int r = 0; r < 8; ++r) {
                float v = acc[mi][ni][r];
                int row = pBase + wm * 32 + mi * 16 + lhalf * 8 + r;
                y[(size_t)row * COUT + cl] = v;
                s += v; s2 += v * v;
            }
        }
        atomicAdd(&ssum[cl], s);
        atomicAdd(&ssumsq[cl], s2);
    }
    __syncthreads();
    if (tid < COUT) {
        atomicAdd(&gsum[tid], ssum[tid]);
        atomicAdd(&gsumsq[tid], ssumsq[tid]);
    }
}

// ---------------- BN finalize: sums -> per-channel scale/shift ----------------
__global__ void bn_finalize_kernel(float* gsum, float* gsumsq,
                                   const float* __restrict__ gamma,
                                   const float* __restrict__ beta) {
    int c = threadIdx.x;
    const float invN = 1.f / (float)NHW;
    float mean = gsum[c] * invN;
    float var  = fmaxf(gsumsq[c] * invN - mean * mean, 0.f);
    float sc   = rsqrtf(var + 1e-5f) * gamma[c];
    gsum[c]   = sc;                    // reuse as scale
    gsumsq[c] = beta[c] - mean * sc;   // reuse as shift
}

// ---------------- BN + ReLU in place ----------------
__global__ void bn_relu_kernel(float4* __restrict__ y,
                               const float* __restrict__ scale,
                               const float* __restrict__ shift, int n4) {
    for (int i = blockIdx.x * blockDim.x + threadIdx.x; i < n4;
         i += gridDim.x * blockDim.x) {
        int c = (i & 63) * 4;          // COUT/4 = 64 vec4 per pixel
        float4 v = y[i];
        v.x = fmaxf(v.x * scale[c + 0] + shift[c + 0], 0.f);
        v.y = fmaxf(v.y * scale[c + 1] + shift[c + 1], 0.f);
        v.z = fmaxf(v.z * scale[c + 2] + shift[c + 2], 0.f);
        v.w = fmaxf(v.w * scale[c + 3] + shift[c + 3], 0.f);
        y[i] = v;
    }
}

extern "C" void kernel_launch(void* const* d_in, const int* in_sizes, int n_in,
                              void* d_out, int out_size, void* d_ws, size_t ws_size,
                              hipStream_t stream) {
    (void)in_sizes; (void)n_in; (void)out_size; (void)ws_size;
    const float* x      = (const float*)d_in[0];   // [N,H,W,CIN]
    const float* kernel = (const float*)d_in[1];   // [KH,KW,CIN,COUT] == [K,COUT]
    const float* gamma  = (const float*)d_in[2];
    const float* beta   = (const float*)d_in[3];
    float* y = (float*)d_out;                      // [NHW, COUT] f32

    // workspace: xq f16 | kqT f16 | {amaxx, amaxk, sum[256], sumsq[256]}
    char* ws = (char*)d_ws;
    _Float16* xq  = (_Float16*)ws;
    _Float16* kqT = (_Float16*)(ws + XQ_BYTES);
    float* stats  = (float*)(ws + XQ_BYTES + KQ_BYTES);
    unsigned* amaxx = (unsigned*)stats;
    unsigned* amaxk = (unsigned*)(stats + 1);
    float* gsum   = stats + 2;
    float* gsumsq = stats + 2 + COUT;

    init_stats_kernel<<<1, 256, 0, stream>>>(amaxx, amaxk, gsum, gsumsq);

    absmax_kernel<<<1024, 256, 0, stream>>>(x, (int)XQ_ELEMS, amaxx);
    absmax_kernel<<<288, 256, 0, stream>>>(kernel, (int)KQ_ELEMS, amaxk);

    quant_x_kernel<<<4096, 256, 0, stream>>>((const float4*)x, (v4h*)xq,
                                             amaxx, (int)(XQ_ELEMS / 4));
    quant_kT_kernel<<<288, 256, 0, stream>>>(kernel, kqT, amaxk);

    conv_wmma_kernel<<<NHW / 64, 256, 0, stream>>>(xq, kqT, y, gsum, gsumsq);

    bn_finalize_kernel<<<1, 256, 0, stream>>>(gsum, gsumsq, gamma, beta);
    bn_relu_kernel<<<8192, 256, 0, stream>>>((float4*)y, gsum, gsumsq,
                                             (int)((size_t)NHW * COUT / 4));
}
